// ConvNeXtBlock_V2_85100482003306
// MI455X (gfx1250) — matrix operand order repaired and checked by reading, not compiled, not run on hardware
//
#include <hip/hip_runtime.h>
#include <hip/hip_bf16.h>

// ---------------------------------------------------------------------------
// ConvNeXt-V2 3D block for MI455X (gfx1250, wave32, WMMA, async-to-LDS)
// B=2, C=128, Ch=512, D=H=W=48, K=7
// ---------------------------------------------------------------------------

#define C_DIM   128
#define CH_DIM  512
#define S_SPA   110592            // 48*48*48
#define N_ROWS  221184            // 2*S_SPA  (divisible by 128)
#define DHW     48
#define HW2     2304              // 48*48

typedef __attribute__((ext_vector_type(16))) _Float16 v16h;
typedef __attribute__((ext_vector_type(8)))  float    v8f;
typedef __attribute__((ext_vector_type(4)))  int      v4i;

#define AS1 __attribute__((address_space(1)))
#define AS3 __attribute__((address_space(3)))

#if __has_builtin(__builtin_amdgcn_global_load_async_to_lds_b128) && \
    __has_builtin(__builtin_amdgcn_s_wait_asynccnt)
#define HAVE_ASYNC 1
__device__ __forceinline__ void async_cp16(const void* gsrc, void* ldst) {
    // GLOBAL_LOAD_ASYNC_TO_LDS_B128 (tracked by ASYNCcnt)
    __builtin_amdgcn_global_load_async_to_lds_b128((AS1 v4i*)gsrc, (AS3 v4i*)ldst, 0, 0);
}
__device__ __forceinline__ void async_wait0() {
    __builtin_amdgcn_s_wait_asynccnt(0);
}
#else
#define HAVE_ASYNC 0
__device__ __forceinline__ void async_wait0() {}
#endif

__device__ __forceinline__ v8f wmma_f16(v16h a, v16h b, v8f c) {
    // D = A(16x32 f16) * B(32x16 f16) + C(16x16 f32)
    return __builtin_amdgcn_wmma_f32_16x16x32_f16(
        /*neg_a=*/false, a, /*neg_b=*/false, b,
        /*c_mod=*/(short)0, c, /*reuse_a=*/false, /*reuse_b=*/false);
}

// A fragment: 16x32 f16 tile, row-major in LDS, row stride `ld` halves.
__device__ __forceinline__ v16h load_a_frag(const _Float16* tile, int ld, int lane) {
    int m  = lane & 15;
    int kh = (lane < 16) ? 0 : 16;
    union { v16h v; uint4 q[2]; } u;
    u.q[0] = *(const uint4*)(tile + m * ld + kh);
    u.q[1] = *(const uint4*)(tile + m * ld + kh + 8);
    return u.v;
}

// B fragment: 32x16 f16 tile stored TRANSPOSED in LDS as Bt[n][k], stride `ld`.
__device__ __forceinline__ v16h load_b_frag(const _Float16* bt, int ld, int lane) {
    int n  = lane & 15;
    int kb = (lane < 16) ? 0 : 16;
    union { v16h v; uint4 q[2]; } u;
    u.q[0] = *(const uint4*)(bt + n * ld + kb);
    u.q[1] = *(const uint4*)(bt + n * ld + kb + 8);
    return u.v;
}

// ---------------------------------------------------------------------------
// Kernel 0: w1 -> fp16 TRANSPOSED [n=512][k=128]; zero GRN accumulators
// ---------------------------------------------------------------------------
__global__ void prep_kernel(const float* __restrict__ w1,
                            _Float16* __restrict__ w1t,
                            float* __restrict__ gx2) {
    int i = blockIdx.x * 256 + threadIdx.x;
    if (i < C_DIM * CH_DIM) {
        int n = i >> 7, k = i & 127;                 // w1t[n][k] = w1[k][n]
        w1t[i] = (_Float16)w1[(size_t)k * CH_DIM + n];
    }
    if (i < 2 * CH_DIM) gx2[i] = 0.0f;
}

// ---------------------------------------------------------------------------
// Kernel 1: depthwise 7x7x7 conv + bias + LayerNorm -> yn[N,128] fp16 chan-last
// ---------------------------------------------------------------------------
__global__ __launch_bounds__(128) void dwln_kernel(
    const float* __restrict__ x, const float* __restrict__ dw_w,
    const float* __restrict__ dw_b, const float* __restrict__ ln_g,
    const float* __restrict__ ln_b, _Float16* __restrict__ yn) {
    int bs = blockIdx.x;
    int b  = (bs >= S_SPA) ? 1 : 0;
    int s  = bs - b * S_SPA;
    int d  = s / HW2;
    int r  = s - d * HW2;
    int hh = r / DHW;
    int ww = r - hh * DHW;
    int c  = threadIdx.x;

    const float* xc = x + (size_t)(b * C_DIM + c) * S_SPA;
    const float* wc = dw_w + c * 343;

    float acc = dw_b[c];
    for (int kd = 0; kd < 7; ++kd) {
        int zd = d + kd - 3;
        if ((unsigned)zd >= (unsigned)DHW) continue;
        for (int kh = 0; kh < 7; ++kh) {
            int zh = hh + kh - 3;
            if ((unsigned)zh >= (unsigned)DHW) continue;
            const float* xrow = xc + zd * HW2 + zh * DHW;
            const float* wrow = wc + (kd * 7 + kh) * 7;
            int zw0 = ww - 3;
#pragma unroll
            for (int kw = 0; kw < 7; ++kw) {
                int zw = zw0 + kw;
                if ((unsigned)zw < (unsigned)DHW) acc += xrow[zw] * wrow[kw];
            }
        }
    }

    __shared__ float s_sum[4], s_sq[4];
    float vsum = acc, vsq = acc * acc;
#pragma unroll
    for (int off = 16; off > 0; off >>= 1) {
        vsum += __shfl_down(vsum, off, 32);
        vsq  += __shfl_down(vsq,  off, 32);
    }
    int wid = c >> 5;
    if ((c & 31) == 0) { s_sum[wid] = vsum; s_sq[wid] = vsq; }
    __syncthreads();
    float tsum = s_sum[0] + s_sum[1] + s_sum[2] + s_sum[3];
    float tsq  = s_sq[0]  + s_sq[1]  + s_sq[2]  + s_sq[3];
    float mu   = tsum * (1.0f / 128.0f);
    float var  = tsq * (1.0f / 128.0f) - mu * mu;
    float vn   = (acc - mu) * rsqrtf(var + 1e-6f) * ln_g[c] + ln_b[c];
    yn[(size_t)bs * C_DIM + c] = (_Float16)vn;
}

// ---------------------------------------------------------------------------
// Kernel 2: GEMM1  h = GELU(yn @ w1 + b1) -> fp16 h[N,512]  + GRN sum(h^2)
// A and B(w1t) tiles are pure contiguous copies -> async global->LDS staging.
// ---------------------------------------------------------------------------
__global__ __launch_bounds__(256) void gemm1_kernel(
    const _Float16* __restrict__ yn, const _Float16* __restrict__ w1t,
    const float* __restrict__ b1, _Float16* __restrict__ h,
    float* __restrict__ gx2) {
    __shared__ _Float16 At[128 * 128];     // [m][k]   32 KB
    __shared__ _Float16 Bt[128 * 128];     // [n][k]   32 KB
    __shared__ float    colsq[128];

    int tid     = threadIdx.x;
    int rowBase = blockIdx.x * 128;
    int colBase = blockIdx.y * 128;

    const _Float16* gA = yn  + (size_t)rowBase * C_DIM;   // 128x128 contiguous
    const _Float16* gB = w1t + (size_t)colBase * C_DIM;   // 128x128 contiguous

#if HAVE_ASYNC
    for (int c = tid; c < 2048; c += 256) {
        async_cp16(gA + c * 8, At + c * 8);
        async_cp16(gB + c * 8, Bt + c * 8);
    }
#else
    {
        const uint4* gA4 = (const uint4*)gA;
        const uint4* gB4 = (const uint4*)gB;
        uint4* sA4 = (uint4*)At;
        uint4* sB4 = (uint4*)Bt;
        for (int i = tid; i < 2048; i += 256) { sA4[i] = gA4[i]; sB4[i] = gB4[i]; }
    }
#endif
    if (tid < 128) colsq[tid] = 0.0f;
    if (rowBase + 128 < N_ROWS)
        __builtin_prefetch(yn + (size_t)(rowBase + 128) * C_DIM + tid * 64, 0, 0);
    async_wait0();
    __syncthreads();

    int w = tid >> 5, lane = tid & 31;
    int wm = w & 3, wn = w >> 2;
    const _Float16* a0b = At + (32 * wm) * 128;
    const _Float16* a1b = At + (32 * wm + 16) * 128;
    const _Float16* bb  = Bt + (64 * wn) * 128;

    v8f acc[2][4] = {};
#pragma unroll
    for (int ks = 0; ks < 128; ks += 32) {
        v16h a0 = load_a_frag(a0b + ks, 128, lane);
        v16h a1 = load_a_frag(a1b + ks, 128, lane);
#pragma unroll
        for (int tn = 0; tn < 4; ++tn) {
            v16h bf = load_b_frag(bb + tn * 16 * 128 + ks, 128, lane);
            acc[0][tn] = wmma_f16(a0, bf, acc[0][tn]);
            acc[1][tn] = wmma_f16(a1, bf, acc[1][tn]);
        }
    }

    int b_idx = (rowBase >= S_SPA) ? 1 : 0;
#pragma unroll
    for (int tm = 0; tm < 2; ++tm) {
#pragma unroll
        for (int tn = 0; tn < 4; ++tn) {
            int colLoc = 64 * wn + 16 * tn + (lane & 15);
            int col    = colBase + colLoc;
            float bias = b1[col];
            float sq   = 0.0f;
#pragma unroll
            for (int r = 0; r < 8; ++r) {
                int row = rowBase + 32 * wm + 16 * tm + ((lane >> 4) << 3) + r;
                float v = acc[tm][tn][r] + bias;
                v = 0.5f * v * (1.0f + erff(v * 0.70710678f));   // exact GELU
                h[(size_t)row * CH_DIM + col] = (_Float16)v;
                sq += v * v;
            }
            atomicAdd(&colsq[colLoc], sq);    // ds_add_f32
        }
    }
    __syncthreads();
    if (tid < 128) atomicAdd(&gx2[b_idx * CH_DIM + colBase + tid], colsq[tid]);
}

// ---------------------------------------------------------------------------
// Kernel 3: GRN scale  s[b,ch] = grn_g[ch] * (gx / (mean(gx)+1e-6)) + 1
// ---------------------------------------------------------------------------
__global__ __launch_bounds__(512) void grn_kernel(const float* __restrict__ gx2,
                                                  const float* __restrict__ grn_g,
                                                  float* __restrict__ sA) {
    int b = blockIdx.x, ch = threadIdx.x;
    float gx = sqrtf(gx2[b * CH_DIM + ch]);
    __shared__ float red[16];
    __shared__ float meanSh;
    float v = gx;
#pragma unroll
    for (int off = 16; off > 0; off >>= 1) v += __shfl_down(v, off, 32);
    if ((ch & 31) == 0) red[ch >> 5] = v;
    __syncthreads();
    if (ch == 0) {
        float t = 0.0f;
        for (int i = 0; i < 16; ++i) t += red[i];
        meanSh = t * (1.0f / 512.0f);
    }
    __syncthreads();
    float nx = gx / (meanSh + 1e-6f);
    sA[b * CH_DIM + ch] = grn_g[ch] * nx + 1.0f;
}

// ---------------------------------------------------------------------------
// Kernel 3b: pack  W2s[b][n][k] = (f16)(s[b,k] * w2[k][n])   (diag-scale fold)
//            bias2eff[c] = b2[c] + sum_ch grn_b[ch]*w2[ch][c]
// ---------------------------------------------------------------------------
__global__ void pack_kernel(const float* __restrict__ w2, const float* __restrict__ sA,
                            const float* __restrict__ grn_b, const float* __restrict__ b2,
                            _Float16* __restrict__ w2s, float* __restrict__ bias2eff) {
    int i = blockIdx.x * 256 + threadIdx.x;
    if (i < 2 * C_DIM * CH_DIM) {
        int b   = i >> 16;
        int rem = i & 65535;
        int n   = rem >> 9;                  // 0..127
        int k   = rem & 511;                 // 0..511
        w2s[i] = (_Float16)(sA[b * CH_DIM + k] * w2[(size_t)k * C_DIM + n]);
    }
    if (blockIdx.x == 0 && threadIdx.x < C_DIM) {
        int c = threadIdx.x;
        float t = b2[c];
        for (int ch = 0; ch < CH_DIM; ++ch) t += grn_b[ch] * w2[(size_t)ch * C_DIM + c];
        bias2eff[c] = t;
    }
}

// ---------------------------------------------------------------------------
// Kernel 4: GEMM2  out = x + h @ W2s[b] + bias2eff   (K=512, double-buffered)
// Both tiles are pure contiguous row copies -> async staging overlaps WMMA.
// ---------------------------------------------------------------------------
__global__ __launch_bounds__(256) void gemm2_kernel(
    const _Float16* __restrict__ h, const _Float16* __restrict__ w2s,
    const float* __restrict__ bias2eff, const float* __restrict__ x,
    float* __restrict__ out) {
    __shared__ _Float16 At[2][128 * 64];   // [m][k]  2 x 16 KB
    __shared__ _Float16 Bt[2][128 * 64];   // [n][k]  2 x 16 KB

    int tid     = threadIdx.x;
    int rowBase = blockIdx.x * 128;
    int b_idx   = (rowBase >= S_SPA) ? 1 : 0;
    int w = tid >> 5, lane = tid & 31;
    int wm = w & 3, wn = w >> 2;

    const _Float16* gA = h   + (size_t)rowBase * CH_DIM;
    const _Float16* gB = w2s + (size_t)b_idx * C_DIM * CH_DIM;

    auto stage = [&](int buf, int k0) {
#if HAVE_ASYNC
        for (int c = tid; c < 1024; c += 256) {
            int rr = c >> 3, p = c & 7;
            async_cp16(gA + (size_t)rr * CH_DIM + k0 + p * 8, &At[buf][rr * 64 + p * 8]);
            async_cp16(gB + (size_t)rr * CH_DIM + k0 + p * 8, &Bt[buf][rr * 64 + p * 8]);
        }
#else
        for (int c = tid; c < 1024; c += 256) {
            int rr = c >> 3, p = c & 7;
            *(uint4*)&At[buf][rr * 64 + p * 8] = *(const uint4*)(gA + (size_t)rr * CH_DIM + k0 + p * 8);
            *(uint4*)&Bt[buf][rr * 64 + p * 8] = *(const uint4*)(gB + (size_t)rr * CH_DIM + k0 + p * 8);
        }
#endif
    };

    v8f acc[2][4] = {};
    stage(0, 0);

    for (int i = 0; i < 8; ++i) {
        int cur = i & 1;
        async_wait0();
        __syncthreads();                       // tiles ready; prev buffer free
        if (i + 1 < 8) stage(cur ^ 1, (i + 1) * 64);

        const _Float16* a0b = &At[cur][(32 * wm) * 64];
        const _Float16* a1b = &At[cur][(32 * wm + 16) * 64];
        const _Float16* bb  = &Bt[cur][(64 * wn) * 64];
#pragma unroll
        for (int kk = 0; kk < 64; kk += 32) {
            v16h a0 = load_a_frag(a0b + kk, 64, lane);
            v16h a1 = load_a_frag(a1b + kk, 64, lane);
#pragma unroll
            for (int tn = 0; tn < 4; ++tn) {
                v16h bf = load_b_frag(bb + tn * 16 * 64 + kk, 64, lane);
                acc[0][tn] = wmma_f16(a0, bf, acc[0][tn]);
                acc[1][tn] = wmma_f16(a1, bf, acc[1][tn]);
            }
        }
    }

    // epilogue: + bias2eff + residual x, write NCDHW fp32
#pragma unroll
    for (int tm = 0; tm < 2; ++tm) {
#pragma unroll
        for (int tn = 0; tn < 4; ++tn) {
            int c    = 64 * wn + 16 * tn + (lane & 15);
            float bv = bias2eff[c];
#pragma unroll
            for (int r = 0; r < 8; ++r) {
                int row  = rowBase + 32 * wm + 16 * tm + ((lane >> 4) << 3) + r;
                int srow = row - b_idx * S_SPA;
                size_t oi = (size_t)(b_idx * C_DIM + c) * S_SPA + srow;
                out[oi] = x[oi] + acc[tm][tn][r] + bv;
            }
        }
    }
}

// ---------------------------------------------------------------------------
// Host launcher
// ---------------------------------------------------------------------------
extern "C" void kernel_launch(void* const* d_in, const int* in_sizes, int n_in,
                              void* d_out, int out_size, void* d_ws, size_t ws_size,
                              hipStream_t stream) {
    (void)in_sizes; (void)n_in; (void)out_size; (void)ws_size;

    const float* x     = (const float*)d_in[0];
    const float* dw_w  = (const float*)d_in[1];
    const float* dw_b  = (const float*)d_in[2];
    const float* ln_g  = (const float*)d_in[3];
    const float* ln_b  = (const float*)d_in[4];
    const float* w1    = (const float*)d_in[5];
    const float* b1    = (const float*)d_in[6];
    const float* grn_g = (const float*)d_in[7];
    const float* grn_b = (const float*)d_in[8];
    const float* w2    = (const float*)d_in[9];
    const float* b2    = (const float*)d_in[10];
    float* out = (float*)d_out;

    // workspace carve-up (all offsets 256B aligned)
    char* ws = (char*)d_ws;
    _Float16* yn   = (_Float16*)(ws);                                  //  56,623,104 B
    _Float16* hbuf = (_Float16*)(ws + 56623104ull);                    // 226,492,416 B
    _Float16* w1t  = (_Float16*)(ws + 56623104ull + 226492416ull);     //     131,072 B
    _Float16* w2s  = (_Float16*)(ws + 56623104ull + 226492416ull + 131072ull);   // 262,144 B
    float*    gx2  = (float*)(ws + 56623104ull + 226492416ull + 393216ull);      //   4,096 B
    float*    sA   = (float*)(ws + 56623104ull + 226492416ull + 397312ull);      //   4,096 B
    float*    b2e  = (float*)(ws + 56623104ull + 226492416ull + 401408ull);      //   512 B

    // 0) transpose+convert w1, zero GRN accumulators
    prep_kernel<<<(C_DIM * CH_DIM + 255) / 256, 256, 0, stream>>>(w1, w1t, gx2);

    // 1) depthwise conv + LN -> yn (fp16, channels-last)
    dwln_kernel<<<N_ROWS, 128, 0, stream>>>(x, dw_w, dw_b, ln_g, ln_b, yn);

    // 2) GEMM1 + GELU + GRN partial sums
    dim3 g1(N_ROWS / 128, CH_DIM / 128);
    gemm1_kernel<<<g1, 256, 0, stream>>>(yn, w1t, b1, hbuf, gx2);

    // 3) GRN scale, then fold scale into W2 + effective bias
    grn_kernel<<<2, 512, 0, stream>>>(gx2, grn_g, sA);
    pack_kernel<<<(2 * C_DIM * CH_DIM + 255) / 256, 256, 0, stream>>>(w2, sA, grn_b, b2, w2s, b2e);

    // 4) GEMM2 + bias + residual (double-buffered async staging)
    gemm2_kernel<<<N_ROWS / 128, 256, 0, stream>>>(hbuf, w2s, b2e, x, out);
}
